// GatedGCN_64579128263346
// MI455X (gfx1250) — matrix operand order repaired and checked by reading, not compiled, not run on hardware
//
#include <hip/hip_runtime.h>

// GatedGCN (ResGatedGraphConv x3 + LeakyReLU + BatchNorm) for MI455X / gfx1250.
// GEMMs via v_wmma_f32_16x16x32_f16 (1 row-tile x NT col-tiles per wave, A reuse);
// edge phase via coalesced gathers + global_atomic_add_f32 scatter; L2-resident.

#define N_NODES   50000
#define N_EDGES   800000
#define ROW_TILES 3125   // 50000 / 16

typedef __attribute__((ext_vector_type(16))) _Float16     v16h;
typedef __attribute__((ext_vector_type(8)))  float        v8f;
typedef __attribute__((ext_vector_type(8)))  unsigned int v8u;

// ---------------------------------------------------------------- utilities
__global__ void to_half_kernel(const float* __restrict__ x,
                               _Float16* __restrict__ xh, long long n) {
  long long i = blockIdx.x * (long long)blockDim.x + threadIdx.x;
  if (i < n) xh[i] = (_Float16)x[i];
}

__global__ void zero_kernel(float* __restrict__ p, int n) {
  int i = blockIdx.x * blockDim.x + threadIdx.x;
  if (i < n) p[i] = 0.0f;
}

// Pack 4 f32 weight matrices W[k][n] (128 x Cout each, k-major) into one
// n-major f16 matrix Wt[4*Cout][128] so each WMMA lane's (k,k+1) pair is a
// contiguous dword. Block order: K | Q | V | Skip.
__global__ void pack_weights_kernel(const float* __restrict__ Wk,
                                    const float* __restrict__ Wq,
                                    const float* __restrict__ Wv,
                                    const float* __restrict__ Ws,
                                    _Float16* __restrict__ Wt,
                                    int Cout, int Cin) {
  int idx = blockIdx.x * blockDim.x + threadIdx.x;
  int tot = 4 * Cout * Cin;
  if (idx >= tot) return;
  int k   = idx % Cin;
  int n   = (idx / Cin) % Cout;
  int blk = idx / (Cin * Cout);
  const float* W = (blk == 0) ? Wk : (blk == 1) ? Wq : (blk == 2) ? Wv : Ws;
  Wt[idx] = (_Float16)W[(unsigned)k * (unsigned)Cout + (unsigned)n];
}

// biasK | biasQ | biasV | conv-bias (skip has no bias; conv bias folds here)
__global__ void pack_bias_kernel(const float* __restrict__ bk,
                                 const float* __restrict__ bq,
                                 const float* __restrict__ bv,
                                 const float* __restrict__ bconv,
                                 float* __restrict__ biasP, int Cout) {
  int i = threadIdx.x + blockIdx.x * blockDim.x;
  if (i >= 4 * Cout) return;
  int blk = i / Cout, n = i % Cout;
  const float* b = (blk == 0) ? bk : (blk == 1) ? bq : (blk == 2) ? bv : bconv;
  biasP[i] = b[n];
}

// ---------------------------------------------------------------- WMMA GEMM
// Y[N][4C] = A_f16[N][Cin] @ Wt_f16[4C][Cin]^T + biasP.
// One wave owns one 16-row tile x NT 16-col tiles; the A fragment is loaded
// once per K-block and feeds NT back-to-back WMMAs (4x less A traffic).
template <int NT>
__global__ void __launch_bounds__(256)
gemm_wmma_kernel(const _Float16* __restrict__ A,
                 const _Float16* __restrict__ Wt,
                 const float* __restrict__ biasP,
                 float* __restrict__ Y,
                 int Cin, int rowTiles, int colGroups) {
  int wave = blockIdx.x * (blockDim.x >> 5) + (threadIdx.x >> 5);
  int numWaves = rowTiles * colGroups;
  if (wave >= numWaves) return;                 // wave-uniform: EXEC stays all-1s
  int rowT = wave / colGroups, colG = wave % colGroups;
  int row0 = rowT * 16, col0 = colG * (16 * NT);
  int ldc  = colGroups * 16 * NT;

  int lane = threadIdx.x & 31;
  int half = lane >> 4;                          // lane group (K-range select)
  int mrel = lane & 15;                          // M for A, N for B/C/D

  const _Float16* Arow = A + (unsigned long long)((unsigned)(row0 + mrel) * (unsigned)Cin);
  const _Float16* Wrow[NT];
#pragma unroll
  for (int t = 0; t < NT; ++t)
    Wrow[t] = Wt + (unsigned long long)((unsigned)(col0 + t * 16 + mrel) * (unsigned)Cin);

  v8f acc[NT] = {};
  for (int kb = 0; kb < Cin; kb += 32) {
    v8u ap{};
#pragma unroll
    for (int r = 0; r < 8; ++r) {
      // 16-bit A 16x32 layout: VGPR r holds K pair {ka, ka+1}
      int ka = kb + ((r < 4) ? (2 * r + 8 * half) : (8 + 2 * r + 8 * half));
      ap[r] = *(const unsigned int*)(Arow + ka);
    }
    v16h av = __builtin_bit_cast(v16h, ap);
#pragma unroll
    for (int t = 0; t < NT; ++t) {
      v8u bp{};
#pragma unroll
      for (int r = 0; r < 8; ++r) {
        // 16-bit B 32x16 layout: VGPR r holds K pair {kbk, kbk+1} (same K order as A), lane = N
        int kbk = kb + ((r < 4) ? (2 * r + 8 * half) : (8 + 2 * r + 8 * half));
        bp[r] = *(const unsigned int*)(Wrow[t] + kbk);
      }
      acc[t] = __builtin_amdgcn_wmma_f32_16x16x32_f16(
          false, av, false, __builtin_bit_cast(v16h, bp),
          (short)0, acc[t], false, false);
    }
  }

#pragma unroll
  for (int t = 0; t < NT; ++t) {
    int cn = col0 + t * 16 + mrel;
    float bias = biasP[cn];
#pragma unroll
    for (int r = 0; r < 8; ++r) {                // C/D layout: VGPR r -> M = r + 8*half
      Y[(unsigned long long)((unsigned)(row0 + r + 8 * half) * (unsigned)ldc) + cn] =
          acc[t][r] + bias;
    }
  }
}

// ---------------------------------------------------------------- edge phase
// eta = sigmoid(k[dst] + q[src] + e@We + be);  atomically Y.skip[dst] += eta*v[src]
template <int C>
__global__ void __launch_bounds__(256)
edge_kernel(const int* __restrict__ ei, const float* __restrict__ eattr,
            const float* __restrict__ We, const float* __restrict__ be,
            float* __restrict__ Y, int E) {
  constexpr int TPE = C / 4;                     // threads per edge (4 ch each)
  unsigned long long gid = blockIdx.x * (unsigned long long)blockDim.x + threadIdx.x;
  if (gid >= (unsigned long long)E * TPE) return;
  unsigned e = (unsigned)(gid / TPE);
  unsigned c = (unsigned)(gid % TPE) * 4u;
  unsigned src = (unsigned)ei[e];
  unsigned dst = (unsigned)ei[(unsigned)E + e];
  float e0 = eattr[3u * e + 0u], e1 = eattr[3u * e + 1u], e2 = eattr[3u * e + 2u];

  constexpr unsigned ld = 4u * C;
  const float* Ysrc = Y + (unsigned long long)(src * ld);  // 32-bit row offsets
  float*       Ydst = Y + (unsigned long long)(dst * ld);

  const float4 kk = *(const float4*)(Ydst + 0u * C + c);
  const float4 qq = *(const float4*)(Ysrc + 1u * C + c);
  const float4 vv = *(const float4*)(Ysrc + 2u * C + c);
  const float4 w0 = *(const float4*)(We + 0u * C + c);
  const float4 w1 = *(const float4*)(We + 1u * C + c);
  const float4 w2 = *(const float4*)(We + 2u * C + c);
  const float4 bb = *(const float4*)(be + c);
  float* aggp = Ydst + 3u * C + c;

  {
    float g = kk.x + qq.x + (e0 * w0.x + e1 * w1.x + e2 * w2.x + bb.x);
    unsafeAtomicAdd(aggp + 0, vv.x / (1.0f + __expf(-g)));
  }
  {
    float g = kk.y + qq.y + (e0 * w0.y + e1 * w1.y + e2 * w2.y + bb.y);
    unsafeAtomicAdd(aggp + 1, vv.y / (1.0f + __expf(-g)));
  }
  {
    float g = kk.z + qq.z + (e0 * w0.z + e1 * w1.z + e2 * w2.z + bb.z);
    unsafeAtomicAdd(aggp + 2, vv.z / (1.0f + __expf(-g)));
  }
  {
    float g = kk.w + qq.w + (e0 * w0.w + e1 * w1.w + e2 * w2.w + bb.w);
    unsafeAtomicAdd(aggp + 3, vv.w / (1.0f + __expf(-g)));
  }
}

// ---------------------------------------------------------------- LReLU + BN
template <int C>
__global__ void __launch_bounds__(256)
bn_stats_kernel(const float* __restrict__ Yagg, int Nn, float* __restrict__ sums) {
  constexpr unsigned ld  = 4u * C;
  constexpr unsigned RPB = 256u / C;             // rows per block iteration
  unsigned c  = threadIdx.x % C;
  unsigned rr = threadIdx.x / C;
  float s = 0.0f, ss = 0.0f;
  for (unsigned r = blockIdx.x * RPB + rr; r < (unsigned)Nn; r += gridDim.x * RPB) {
    float v = Yagg[(unsigned long long)(r * ld) + c];
    v = (v >= 0.0f) ? v : 0.01f * v;             // LeakyReLU before stats
    s += v; ss += v * v;
  }
  unsafeAtomicAdd(&sums[c], s);
  unsafeAtomicAdd(&sums[C + c], ss);
}

__global__ void bn_finalize_kernel(const float* __restrict__ sums,
                                   const float* __restrict__ gamma,
                                   const float* __restrict__ beta,
                                   float* __restrict__ affine, int C, float invN) {
  int c = threadIdx.x;
  if (c >= C) return;
  float mean = sums[c] * invN;
  float var  = sums[C + c] * invN - mean * mean;
  float a = gamma[c] * rsqrtf(var + 1e-5f);
  affine[c]     = a;
  affine[C + c] = beta[c] - mean * a;
}

template <int C, bool F16OUT, bool F32OUT>
__global__ void __launch_bounds__(256)
bn_apply_kernel(const float* __restrict__ Yagg, int Nn,
                const float* __restrict__ affine,
                _Float16* __restrict__ xh, float* __restrict__ out) {
  constexpr unsigned ld = 4u * C;
  unsigned idx = blockIdx.x * blockDim.x + threadIdx.x;   // < 6.4M, fits u32
  if (idx >= (unsigned)Nn * (unsigned)C) return;
  unsigned c = idx % C;
  unsigned r = idx / C;
  float v = Yagg[(unsigned long long)(r * ld) + c];
  v = (v >= 0.0f) ? v : 0.01f * v;               // LeakyReLU
  float y = affine[c] * v + affine[C + c];
  if (F16OUT) xh[idx]  = (_Float16)y;
  if (F32OUT) out[idx] = y;
}

// ---------------------------------------------------------------- launcher
extern "C" void kernel_launch(void* const* d_in, const int* in_sizes, int n_in,
                              void* d_out, int out_size, void* d_ws, size_t ws_size,
                              hipStream_t stream) {
  const float* x     = (const float*)d_in[0];
  const int*   ei    = (const int*)  d_in[1];
  const float* eattr = (const float*)d_in[3];   // d_in[2] = batch (unused)

  char* ws = (char*)d_ws;
  size_t off = 0;
  auto carve = [&](size_t bytes) -> char* {
    char* p = ws + off;
    off = (off + bytes + 255) & ~(size_t)255;
    return p;
  };
  float*    Y      = (float*)   carve((size_t)N_NODES * 512 * sizeof(float));
  _Float16* Xh     = (_Float16*)carve((size_t)N_NODES * 128 * sizeof(_Float16));
  _Float16* Wt0    = (_Float16*)carve((size_t)512 * 128 * sizeof(_Float16));
  _Float16* Wt1    = (_Float16*)carve((size_t)512 * 128 * sizeof(_Float16));
  _Float16* Wt2    = (_Float16*)carve((size_t)32  * 128 * sizeof(_Float16));
  float*    bP0    = (float*)   carve(512 * sizeof(float));
  float*    bP1    = (float*)   carve(512 * sizeof(float));
  float*    bP2    = (float*)   carve(32  * sizeof(float));
  float*    bnSums = (float*)   carve(256 * sizeof(float));
  float*    affine = (float*)   carve(256 * sizeof(float));
  _Float16* WtL[3] = {Wt0, Wt1, Wt2};
  float*    bPL[3] = {bP0, bP1, bP2};

  {   // x -> f16 staging for the first GEMM
    long long n = (long long)N_NODES * 128;
    to_half_kernel<<<(int)((n + 255) / 256), 256, 0, stream>>>(x, Xh, n);
  }

  const int pbase[3] = {4, 16, 28};
  const int nbase[3] = {14, 26, 38};
  const int CoutA[3] = {128, 128, 8};

  for (int l = 0; l < 3; ++l) {
    int pb = pbase[l];
    const float* Wk  = (const float*)d_in[pb + 0];
    const float* bk  = (const float*)d_in[pb + 1];
    const float* Wq  = (const float*)d_in[pb + 2];
    const float* bq  = (const float*)d_in[pb + 3];
    const float* Wv  = (const float*)d_in[pb + 4];
    const float* bv  = (const float*)d_in[pb + 5];
    const float* We  = (const float*)d_in[pb + 6];
    const float* be  = (const float*)d_in[pb + 7];
    const float* Wsk = (const float*)d_in[pb + 8];
    const float* bcv = (const float*)d_in[pb + 9];
    const float* gamma = (const float*)d_in[nbase[l] + 0];
    const float* beta  = (const float*)d_in[nbase[l] + 1];

    const int C = CoutA[l], Cin = 128;

    int tot = 4 * C * Cin;
    pack_weights_kernel<<<(tot + 255) / 256, 256, 0, stream>>>(Wk, Wq, Wv, Wsk,
                                                               WtL[l], C, Cin);
    pack_bias_kernel<<<1, 4 * C, 0, stream>>>(bk, bq, bv, bcv, bPL[l], C);

    if (C == 128) {
      // 32 col tiles -> 8 groups of 4; 3125*8 = 25000 waves
      int numWaves = ROW_TILES * 8;
      gemm_wmma_kernel<4><<<(numWaves + 7) / 8, 256, 0, stream>>>(
          Xh, WtL[l], bPL[l], Y, Cin, ROW_TILES, 8);
      long long nt = (long long)N_EDGES * 32;
      edge_kernel<128><<<(int)((nt + 255) / 256), 256, 0, stream>>>(ei, eattr, We,
                                                                    be, Y, N_EDGES);
      zero_kernel<<<1, 256, 0, stream>>>(bnSums, 2 * C);
      bn_stats_kernel<128><<<256, 256, 0, stream>>>(Y + 3 * C, N_NODES, bnSums);
    } else {
      // 2 col tiles -> 1 group of 2; 3125 waves
      int numWaves = ROW_TILES * 1;
      gemm_wmma_kernel<2><<<(numWaves + 7) / 8, 256, 0, stream>>>(
          Xh, WtL[l], bPL[l], Y, Cin, ROW_TILES, 1);
      long long nt = (long long)N_EDGES * 2;
      edge_kernel<8><<<(int)((nt + 255) / 256), 256, 0, stream>>>(ei, eattr, We,
                                                                  be, Y, N_EDGES);
      zero_kernel<<<1, 256, 0, stream>>>(bnSums, 2 * C);
      bn_stats_kernel<8><<<256, 256, 0, stream>>>(Y + 3 * C, N_NODES, bnSums);
    }

    bn_finalize_kernel<<<1, C, 0, stream>>>(bnSums, gamma, beta, affine, C,
                                            1.0f / (float)N_NODES);

    long long ne = (long long)N_NODES * C;
    if (l < 2)
      bn_apply_kernel<128, true, false><<<(int)((ne + 255) / 256), 256, 0, stream>>>(
          Y + 3 * C, N_NODES, affine, Xh, nullptr);
    else
      bn_apply_kernel<8, false, true><<<(int)((ne + 255) / 256), 256, 0, stream>>>(
          Y + 3 * C, N_NODES, affine, nullptr, (float*)d_out);
  }
}
